// RoutedMlpExperts_51256139710801
// MI455X (gfx1250) — matrix operand order repaired and checked by reading, not compiled, not run on hardware
//
#include <hip/hip_runtime.h>

// ---------------------------------------------------------------------------
// RoutedMlpExperts on gfx1250 (MI455X): bf16 WMMA + TDM-staged LDS pipeline
//   tokens = B*S = 8192, D = 2048, E*R = 4096
//   GEMM1: H[8192x4096] = X * Wup^T   (+ fused silu * expert_mask, bf16 out)
//   GEMM2: Out[8192x2048] = H * Wdn^T (f32 out)
// Block tile 64M x 256N, 8 waves (2x4), wave tile 32M x 64N (8 accumulators).
// B tile (256 rows x 32 K, bf16) staged into LDS by the Tensor Data Mover,
// double buffered, padded rows (80B) for bank-conflict-free ds_load_b128.
// ---------------------------------------------------------------------------

typedef __attribute__((ext_vector_type(4)))  __bf16 bf16x4;
typedef __attribute__((ext_vector_type(8)))  __bf16 bf16x8;
typedef __attribute__((ext_vector_type(16))) __bf16 bf16x16;
typedef __attribute__((ext_vector_type(8)))  float  floatx8;
typedef unsigned int u32x4 __attribute__((ext_vector_type(4)));
typedef int          i32x8 __attribute__((ext_vector_type(8)));
typedef int          i32x4 __attribute__((ext_vector_type(4)));

#define TOKENS  8192
#define EMBED   2048
#define EXPERTS 64
#define NER     4096
#define LDSROW  40        // padded LDS row stride in bf16 elems (80 B = 5x16B)

#if defined(__has_builtin)
#if __has_builtin(__builtin_amdgcn_tensor_load_to_lds)
#define HAVE_TDM 1
#endif
#endif
#ifndef HAVE_TDM
#define HAVE_TDM 0
#endif

union ABFrag {
  bf16x16 full;
  bf16x8  h8[2];
};

// ---------------------------------------------------------------------------
// f32 -> bf16 conversion (vectorized x4)
// ---------------------------------------------------------------------------
__global__ __launch_bounds__(256) void cvt_f32_to_bf16(const float* __restrict__ in,
                                                       __bf16* __restrict__ out,
                                                       int n4) {
  int i = blockIdx.x * blockDim.x + threadIdx.x;
  if (i >= n4) return;
  float4 v = ((const float4*)in)[i];
  bf16x4 o;
  o.x = (__bf16)v.x;
  o.y = (__bf16)v.y;
  o.z = (__bf16)v.z;
  o.w = (__bf16)v.w;
  ((bf16x4*)out)[i] = o;
}

// ---------------------------------------------------------------------------
// A fragment (16x32 bf16): lane<16 -> row M=lane, K [k0..k0+15];
// lane>=16 -> K [k0+16..k0+31] (contiguous, matches B fragment).
// ---------------------------------------------------------------------------
__device__ __forceinline__ bf16x16 load_a_frag(const __bf16* row, int k0, int hf) {
  ABFrag a;
  a.h8[0] = *(const bf16x8*)(row + k0 + hf * 16);
  a.h8[1] = *(const bf16x8*)(row + k0 + hf * 16 + 8);
  return a.full;
}

// ---------------------------------------------------------------------------
// TDM: issue a 2D tensor_load_to_lds for a 256-row x 32-elem bf16 tile.
// D# packing per CDNA5 ISA:
//   group0: count=1 | lds_addr | global_addr[56:0] | type=2
//   group1: data_size=2B, pad_enable, pad_interval=16 DW (64B),
//           pad_amount=4 DW (16B) -> padded LDS row stride 80B;
//           tensor_dim0=K, tensor_dim1=N, tile_dim0=32, tile_dim1=256,
//           tensor_dim0_stride=K
// 6-arg builtin (clang-23 / therock-10.0 toolchain):
//   (uint32x4 g0, int32x8 g1, int32x4 g2, int32x4 g3, int32x8 pad, i32 cpol)
// ---------------------------------------------------------------------------
#if HAVE_TDM
__device__ __forceinline__ void tdm_load_tile(const __bf16* gsrc, void* ldst,
                                              int K_, int N_) {
  unsigned long long ga = (unsigned long long)gsrc;
  u32x4 g0;
  g0[0] = 1u;                                            // count=1, user desc
  g0[1] = (unsigned)(unsigned long long)ldst;            // LDS byte address
  g0[2] = (unsigned)ga;                                  // global_addr[31:0]
  g0[3] = (unsigned)((ga >> 32) & 0x01FFFFFFu) | (2u << 30);  // [56:32] | type=2

  i32x8 g1;
  g1[0] = (int)((1u << 16)    // data_size = 2 bytes
              | (1u << 20)    // pad_enable
              | (3u << 22)    // pad_interval: 16 DWORDs (=64B rows)
              | (3u << 25));  // pad_amount: 4 DWORDs (=16B)
  g1[1] = (int)(((unsigned)K_ & 0xFFFFu) << 16);                     // dim0 lo16
  g1[2] = (int)(((unsigned)K_ >> 16) | (((unsigned)N_ & 0xFFFFu) << 16)); // dim0 hi | dim1 lo
  g1[3] = (int)(((unsigned)N_ >> 16) | (32u << 16));                 // dim1 hi | tile_dim0=32
  g1[4] = 256;                                                       // tile_dim1=256
  g1[5] = K_;                                                        // dim0_stride lo32
  g1[6] = 0;
  g1[7] = 0;

  i32x4 z4 = {};
  i32x8 z8 = {};
  __builtin_amdgcn_tensor_load_to_lds(g0, g1, z4, z4, z8, 0);
}
#endif

// Stage one B tile (rows nBlk..nBlk+255, k chunk k0) into LDS buffer `buf`.
__device__ __forceinline__ void stage_tile(const __bf16* __restrict__ Wb, int nBlk,
                                           int k0, __bf16* buf, int K_, int N_) {
#if HAVE_TDM
  if ((threadIdx.x >> 5) == 0)           // one wave drives the TDM
    tdm_load_tile(Wb + (size_t)nBlk * K_ + k0, buf, K_, N_);
#else
  // fallback: thread t copies row t (32 bf16 = 64B) into padded LDS row
  const __bf16* src = Wb + (size_t)(nBlk + threadIdx.x) * K_ + k0;
  __bf16* dst = buf + threadIdx.x * LDSROW;
#pragma unroll
  for (int c = 0; c < 4; ++c)
    *(bf16x8*)(dst + c * 8) = *(const bf16x8*)(src + c * 8);
#endif
}

__device__ __forceinline__ void wait_tile_notlast() {
#if HAVE_TDM
  __builtin_amdgcn_s_wait_tensorcnt(1);  // cur tile complete (next in flight)
#endif
}
__device__ __forceinline__ void wait_tile_last() {
#if HAVE_TDM
  __builtin_amdgcn_s_wait_tensorcnt(0);
#endif
}

// ---------------------------------------------------------------------------
// Fused GEMM: out[M=8192][N] = A[8192][K](bf16) * Wb[N][K]^T(bf16)
//   SILU=true : out = bf16( silu(acc) * mask[token, col/64] )  -> outH
//   SILU=false: out = f32(acc)                                 -> outF
// Grid: (N/256, 8192/64), block 256.
// ---------------------------------------------------------------------------
template <int K, int N, bool SILU>
__global__ __launch_bounds__(256) void gemm_bf16_tdm(
    const __bf16* __restrict__ Aab,   // [TOKENS][K]
    const __bf16* __restrict__ Wb,    // [N][K]
    const float*  __restrict__ mask,  // [TOKENS][EXPERTS] (SILU only)
    __bf16* __restrict__ outH,        // [TOKENS][N] (SILU)
    float*  __restrict__ outF)        // [TOKENS][N] (!SILU)
{
  __shared__ __bf16 Bs[2][256 * LDSROW];   // 2 x 20 KB, padded rows

  const int w    = threadIdx.x >> 5;
  const int lane = threadIdx.x & 31;
  const int lr   = lane & 15;
  const int hf   = lane >> 4;

  const int mBlk = blockIdx.y * 64;
  const int nBlk = blockIdx.x * 256;
  const int m0   = mBlk + (w >> 2) * 32;   // wave M base (2 tiles: +0, +16)
  const int nW   = (w & 3) * 64;           // wave N base within block tile
  const int n0   = nBlk + nW;

  const __bf16* arow0 = Aab + (size_t)(m0 + lr) * K;
  const __bf16* arow1 = Aab + (size_t)(m0 + 16 + lr) * K;

  floatx8 acc[2][4] = {};

  constexpr int NSTEPS = K / 32;

  // stage step 0 B tile into buffer 0
  stage_tile(Wb, nBlk, 0, &Bs[0][0], K, N);

  // preload A fragments for step 0
  bf16x16 a0 = load_a_frag(arow0, 0, hf);
  bf16x16 a1 = load_a_frag(arow1, 0, hf);

  for (int step = 0; step < NSTEPS; ++step) {
    const int cur = step & 1;
    const bool notlast = (step + 1 < NSTEPS);

    // issue next tile DMA + next A fragment loads (overlap with compute)
    bf16x16 na0 = a0, na1 = a1;
    if (notlast) {
      stage_tile(Wb, nBlk, (step + 1) * 32, &Bs[cur ^ 1][0], K, N);
      __builtin_prefetch(arow0 + (step + 2) * 32, 0, 1);
      na0 = load_a_frag(arow0, (step + 1) * 32, hf);
      na1 = load_a_frag(arow1, (step + 1) * 32, hf);
    }

    // wait for current tile's DMA, then rendezvous
    if (notlast) wait_tile_notlast();
    else         wait_tile_last();
    __syncthreads();

    // B fragment (32x16 bf16): lane<16 -> col N=lane, K [0..15];
    // lane>=16 -> same cols, K [16..31]; rows are padded LDS rows.
    const __bf16* bbase = &Bs[cur][0];
#pragma unroll
    for (int t = 0; t < 4; ++t) {
      bf16x16 bfrag = *(const bf16x16*)(bbase + (nW + t * 16 + lr) * LDSROW + hf * 16);
      acc[0][t] = __builtin_amdgcn_wmma_f32_16x16x32_bf16(
          false, a0, false, bfrag, (short)0, acc[0][t], false, false);
      acc[1][t] = __builtin_amdgcn_wmma_f32_16x16x32_bf16(
          false, a1, false, bfrag, (short)0, acc[1][t], false, false);
    }
    __syncthreads();   // all reads of Bs[cur] done before it is overwritten

    a0 = na0;
    a1 = na1;
  }

  // ------------------------- epilogue -------------------------
  // C/D layout: lane -> col offset lr; VGPR j -> row (j + hf*8)
  if (SILU) {
    const int e = n0 >> 6;   // R==64, n0 is 64-aligned -> one expert per wave
#pragma unroll
    for (int mt = 0; mt < 2; ++mt) {
      float mk[8];
#pragma unroll
      for (int j = 0; j < 8; ++j)
        mk[j] = mask[(size_t)(m0 + mt * 16 + j + hf * 8) * EXPERTS + e];
#pragma unroll
      for (int t = 0; t < 4; ++t) {
        const int col = n0 + t * 16 + lr;
#pragma unroll
        for (int j = 0; j < 8; ++j) {
          float v = acc[mt][t][j];
          v = v / (1.0f + __expf(-v));   // silu
          v *= mk[j];
          outH[(size_t)(m0 + mt * 16 + j + hf * 8) * N + col] = (__bf16)v;
        }
      }
    }
  } else {
#pragma unroll
    for (int mt = 0; mt < 2; ++mt) {
#pragma unroll
      for (int t = 0; t < 4; ++t) {
        const int col = n0 + t * 16 + lr;
#pragma unroll
        for (int j = 0; j < 8; ++j)
          outF[(size_t)(m0 + mt * 16 + j + hf * 8) * N + col] = acc[mt][t][j];
      }
    }
  }
}

// ---------------------------------------------------------------------------
// Host launcher
// Inputs: x[4,2048,2048] f32, expert_mask[4,2048,64] f32,
//         w_up[64,64,2048] f32, w_down[2048,64,64] f32 ; output f32 x-shaped.
// Workspace (128 MB): Xb @0 (32MB), Wub @32MB (16MB), Wdb @48MB (16MB),
//                     Hb @64MB (64MB)
// ---------------------------------------------------------------------------
extern "C" void kernel_launch(void* const* d_in, const int* in_sizes, int n_in,
                              void* d_out, int out_size, void* d_ws, size_t ws_size,
                              hipStream_t stream) {
  (void)in_sizes; (void)n_in; (void)out_size; (void)ws_size;

  const float* x    = (const float*)d_in[0];
  const float* mask = (const float*)d_in[1];
  const float* w_up = (const float*)d_in[2];
  const float* w_dn = (const float*)d_in[3];
  float* out = (float*)d_out;

  char* ws = (char*)d_ws;
  __bf16* Xb  = (__bf16*)(ws);
  __bf16* Wub = (__bf16*)(ws + ((size_t)32 << 20));
  __bf16* Wdb = (__bf16*)(ws + ((size_t)48 << 20));
  __bf16* Hb  = (__bf16*)(ws + ((size_t)64 << 20));

  {
    int n4 = (TOKENS * EMBED) / 4;
    cvt_f32_to_bf16<<<n4 / 256, 256, 0, stream>>>(x, Xb, n4);
  }
  {
    int n4 = (NER * EMBED) / 4;
    cvt_f32_to_bf16<<<n4 / 256, 256, 0, stream>>>(w_up, Wub, n4);
  }
  {
    int n4 = (EMBED * NER) / 4;
    cvt_f32_to_bf16<<<n4 / 256, 256, 0, stream>>>(w_dn, Wdb, n4);
  }

  // GEMM1: up-proj + silu + expert mask -> Hb (bf16)
  gemm_bf16_tdm<EMBED, NER, true>
      <<<dim3(NER / 256, TOKENS / 64), 256, 0, stream>>>(Xb, Wub, mask, Hb, nullptr);

  // GEMM2: down-proj -> out (f32)
  gemm_bf16_tdm<NER, EMBED, false>
      <<<dim3(EMBED / 256, TOKENS / 64), 256, 0, stream>>>(Hb, Wdb, nullptr, nullptr, out);
}
